// MultiHeadAttention_38963943309396
// MI455X (gfx1250) — matrix operand order repaired and checked by reading, not compiled, not run on hardware
//
#include <hip/hip_runtime.h>

// ---------------------------------------------------------------------------
// MI455X (gfx1250) multi-head attention, bf16 WMMA (v_wmma_f32_16x16x32_bf16)
//   B=2, T=2048, H=8, K=128 (head dim == model dim)
// Pipeline:
//   k1: QKV projection (f32 -> bf16), per-head layout, Q pre-scaled 1/sqrt(128)
//   k2: flash attention per (b,h), online softmax in-registers, bf16 WMMA,
//       Q/V tiles staged via GLOBAL_LOAD_ASYNC_TO_LDS (ASYNCcnt path)
//   k3: output projection y @ Wu + bu -> f32
// ---------------------------------------------------------------------------

typedef __attribute__((ext_vector_type(16))) __bf16 v16bf;
typedef __attribute__((ext_vector_type(8)))  float  v8f;

#define TT    2048
#define HH    8
#define BB    2
#define DK    128
#define MROWS (BB * TT)              // 4096
#define QSCALE 0.08838834764831845f // 1/sqrt(128)

#ifndef __has_builtin
#define __has_builtin(x) 0
#endif

// ---- CDNA5 async global->LDS path (ASYNCcnt), with safe fallback ----------
#if defined(__gfx1250__) && __has_builtin(__builtin_amdgcn_global_load_async_to_lds_b128)
#define HAVE_ASYNC_LDS 1
#else
#define HAVE_ASYNC_LDS 0
#endif

#if HAVE_ASYNC_LDS
typedef int v4i __attribute__((vector_size(16)));
typedef __attribute__((address_space(1))) v4i* as1_v4i_ptr;   // global v4i*
typedef __attribute__((address_space(3))) v4i* as3_v4i_ptr;   // LDS v4i*
#endif

__device__ __forceinline__ void async_copy_b128(void* lds_dst, const void* gsrc) {
#if HAVE_ASYNC_LDS
  __builtin_amdgcn_global_load_async_to_lds_b128(
      (as1_v4i_ptr)gsrc, (as3_v4i_ptr)lds_dst, 0, 0);
#else
  *reinterpret_cast<uint4*>(lds_dst) = *reinterpret_cast<const uint4*>(gsrc);
#endif
}

__device__ __forceinline__ void async_wait0() {
#if HAVE_ASYNC_LDS
#if __has_builtin(__builtin_amdgcn_s_wait_asynccnt)
  __builtin_amdgcn_s_wait_asynccnt(0);
#else
  asm volatile("s_wait_asynccnt 0x0" ::: "memory");
#endif
#endif
}

// ---- WMMA wrapper ---------------------------------------------------------
__device__ __forceinline__ v8f wmma_bf16(v16bf a, v16bf b, v8f c) {
  // (neg_a, A, neg_b, B, c_mod, C, reuse_a, reuse_b)
  return __builtin_amdgcn_wmma_f32_16x16x32_bf16(false, a, false, b,
                                                 (short)0, c, false, false);
}

// ---- Fragment loaders (LDS row stride fixed at 128 elements) --------------
// A (16x32 bf16): lane L holds row M=L&15; lanes 0-15: K = {0..7,16..23},
// lanes 16-31: K = {8..15,24..31}. Two contiguous 16B chunks per lane.
__device__ __forceinline__ v16bf frag_a(const __bf16* p, int lane) {
  const __bf16* r = p + (lane & 15) * 128 + ((lane >> 4) << 3);
  v16bf a;
#pragma unroll
  for (int i = 0; i < 8; ++i) a[i] = r[i];
#pragma unroll
  for (int i = 0; i < 8; ++i) a[i + 8] = r[16 + i];
  return a;
}

// B (32x16 bf16): lane L holds column N=L&15; lanes 0-15: K = {0..7,16..23},
// lanes 16-31: K = {8..15,24..31} (same K order as frag_a).
__device__ __forceinline__ v16bf frag_b(const __bf16* p, int lane) {
  const __bf16* r = p + (lane & 15) + ((lane >> 4) << 3) * 128;
  v16bf b;
#pragma unroll
  for (int i = 0; i < 8; ++i) b[i] = r[i * 128];
#pragma unroll
  for (int i = 0; i < 8; ++i) b[i + 8] = r[(16 + i) * 128];
  return b;
}

// ===========================================================================
// Kernel 1: fused QKV projection.
//   grid = (32 row-tiles, 24 col-tiles): colTile/8 selects {Wq,Wk,Wv}.
//   Output layout: [b][h][t][128] bf16, Q scaled by 1/sqrt(128).
//   LDS: sX 128x128 bf16 + sW 128x128 bf16 = 64 KB dynamic.
// ===========================================================================
__global__ __launch_bounds__(256) void qkv_proj_kernel(
    const float* __restrict__ x, const float* __restrict__ Wq,
    const float* __restrict__ Wk, const float* __restrict__ Wv,
    __bf16* __restrict__ Qo, __bf16* __restrict__ Ko, __bf16* __restrict__ Vo) {
  extern __shared__ char smem[];
  __bf16* sX = reinterpret_cast<__bf16*>(smem);            // 128*128
  __bf16* sW = sX + 128 * 128;                             // 128*128

  const int tid  = threadIdx.x;
  const int lane = tid & 31, w = tid >> 5;
  const int rowBase = blockIdx.x * 128;
  const int my  = blockIdx.y;
  const int mat = my >> 3;          // 0=Q 1=K 2=V
  const int h   = my & 7;
  const int c0  = h * 128;

  const float* W   = (mat == 0) ? Wq : (mat == 1) ? Wk : Wv;
  __bf16*      dst = (mat == 0) ? Qo : (mat == 1) ? Ko : Vo;
  const float  osc = (mat == 0) ? QSCALE : 1.0f;

  // Stage x tile (f32 -> bf16), coalesced float4 reads.
  const float4* xv = reinterpret_cast<const float4*>(x + (size_t)rowBase * DK);
  for (int i = tid; i < 4096; i += 256) {
    float4 f = xv[i];
    int row = i >> 5, c4 = (i & 31) << 2;
    __bf16* d = &sX[row * 128 + c4];
    d[0] = (__bf16)f.x; d[1] = (__bf16)f.y; d[2] = (__bf16)f.z; d[3] = (__bf16)f.w;
  }
  // Stage W tile [k=0..127][c0..c0+127].
  for (int i = tid; i < 4096; i += 256) {
    int kr = i >> 5, c4 = (i & 31) << 2;
    float4 f = *reinterpret_cast<const float4*>(W + (size_t)kr * (DK * HH) + c0 + c4);
    __bf16* d = &sW[kr * 128 + c4];
    d[0] = (__bf16)f.x; d[1] = (__bf16)f.y; d[2] = (__bf16)f.z; d[3] = (__bf16)f.w;
  }
  __syncthreads();

  // 8 waves as 4(M) x 2(N): each wave -> 32x64 sub-tile = 2x4 WMMA tiles.
  const int mBase = (w >> 1) * 32, nBase = (w & 1) * 64;
  v8f acc[2][4] = {};
#pragma unroll
  for (int kk = 0; kk < 128; kk += 32) {
    v16bf a0 = frag_a(sX + mBase * 128 + kk, lane);
    v16bf a1 = frag_a(sX + (mBase + 16) * 128 + kk, lane);
#pragma unroll
    for (int j = 0; j < 4; ++j) {
      v16bf bf = frag_b(sW + kk * 128 + nBase + 16 * j, lane);
      acc[0][j] = wmma_bf16(a0, bf, acc[0][j]);
      acc[1][j] = wmma_bf16(a1, bf, acc[1][j]);
    }
  }

  // Store into per-head [b][h][t][128] bf16.
  const int half = lane >> 4, nc = lane & 15;
#pragma unroll
  for (int i = 0; i < 2; ++i)
#pragma unroll
    for (int j = 0; j < 4; ++j)
#pragma unroll
      for (int r = 0; r < 8; ++r) {
        int gr  = rowBase + mBase + 16 * i + r + 8 * half;   // 0..4095
        int bb  = gr >> 11, t = gr & (TT - 1);
        int col = nBase + 16 * j + nc;                       // 0..127
        size_t o = ((size_t)(bb * HH + h) * TT + t) * DK + col;
        dst[o] = (__bf16)(acc[i][j][r] * osc);
      }
}

// ===========================================================================
// Kernel 2: flash attention. grid = (T/128 q-tiles, B*H). 8 waves/block,
// wave w owns query rows [w*16, w*16+16) x all 128 keys of the tile, so the
// online softmax reduces within 16-lane halves via __shfl_xor.
// Q and V tiles staged with async global->LDS; K tile transposed via VGPRs.
// LDS: sQ + sKT + sV + sP = 4 * 32 KB = 128 KB dynamic.
// ===========================================================================
__global__ __launch_bounds__(256) void flash_attn_kernel(
    const __bf16* __restrict__ Q, const __bf16* __restrict__ Km,
    const __bf16* __restrict__ V, __bf16* __restrict__ Y) {
  extern __shared__ char smem[];
  __bf16* sQ  = reinterpret_cast<__bf16*>(smem);   // [128][128] q rows
  __bf16* sKT = sQ  + 128 * 128;                   // [feat][key] (transposed)
  __bf16* sV  = sKT + 128 * 128;                   // [key][feat]
  __bf16* sP  = sV  + 128 * 128;                   // 8 waves x [16][128]

  const int tid = threadIdx.x, lane = tid & 31, w = tid >> 5;
  const int bh = blockIdx.y;                 // b*8 + h
  const int b  = bh >> 3, h = bh & 7;
  const int qRowBase = blockIdx.x * 128;

  const __bf16* Qh = Q  + (size_t)bh * TT * DK;
  const __bf16* Kh = Km + (size_t)bh * TT * DK;
  const __bf16* Vh = V  + (size_t)bh * TT * DK;

  // Resident Q tile: async global->LDS, 16B per lane per op.
  {
    const __bf16* src = Qh + (size_t)qRowBase * DK + tid * 8;
    __bf16* dst = sQ + tid * 8;
#pragma unroll
    for (int i = 0; i < 8; ++i)
      async_copy_b128(dst + i * 2048, src + i * 2048);   // 2048 elems = 4 KB
  }

  const int m0 = w * 16;                     // wave's query rows within tile
  const int half = lane >> 4, nc = lane & 15;
  float mrow[8], lrow[8];
#pragma unroll
  for (int r = 0; r < 8; ++r) { mrow[r] = -1e30f; lrow[r] = 0.0f; }
  v8f oacc[8] = {};

  for (int kb = 0; kb < TT / 128; ++kb) {
    __syncthreads();   // all waves done reading previous sKT/sV
    // V tile: async straight copy.
    {
      const __bf16* vsrc = Vh + (size_t)kb * 128 * DK + tid * 8;
      __bf16* vdst = sV + tid * 8;
#pragma unroll
      for (int i = 0; i < 8; ++i)
        async_copy_b128(vdst + i * 2048, vsrc + i * 2048);
    }
    // K tile transposed into sKT[feat][key]: 16B global reads, b16 LDS scatter.
    {
      const __bf16* Kt = Kh + (size_t)kb * 128 * DK;
      for (int i = tid; i < 2048; i += 256) {
        int kr = i >> 4, kf8 = (i & 15) << 3;
        uint4 raw = *reinterpret_cast<const uint4*>(Kt + kr * DK + kf8);
        const __bf16* e = reinterpret_cast<const __bf16*>(&raw);
#pragma unroll
        for (int j = 0; j < 8; ++j) sKT[(kf8 + j) * 128 + kr] = e[j];
      }
    }
    // Prefetch next K/V tiles into GL2 while this iteration computes.
    if (kb + 1 < TT / 128) {
      __builtin_prefetch(Kh + (size_t)(kb + 1) * 128 * DK + tid * 64, 0, 1);
      __builtin_prefetch(Vh + (size_t)(kb + 1) * 128 * DK + tid * 64, 0, 1);
    }
    async_wait0();     // wave's async tile loads landed in LDS
    __syncthreads();   // everyone's tiles visible

    // S = Q_tile @ K_tile^T   (16 x 128 per wave, 8 N-tiles)
    v8f sacc[8] = {};
#pragma unroll
    for (int kk = 0; kk < 128; kk += 32) {
      v16bf a = frag_a(sQ + m0 * 128 + kk, lane);
#pragma unroll
      for (int j = 0; j < 8; ++j) {
        v16bf bf = frag_b(sKT + kk * 128 + 16 * j, lane);
        sacc[j] = wmma_bf16(a, bf, sacc[j]);
      }
    }

    // Online softmax: per-row (VGPR index = row, half-wave = row+8).
    float pscale[8], tsum[8];
#pragma unroll
    for (int r = 0; r < 8; ++r) {
      float m = sacc[0][r];
#pragma unroll
      for (int j = 1; j < 8; ++j) m = fmaxf(m, sacc[j][r]);
#pragma unroll
      for (int d = 1; d < 16; d <<= 1) m = fmaxf(m, __shfl_xor(m, d, 32));
      float mn = fmaxf(mrow[r], m);
      pscale[r] = __expf(mrow[r] - mn);
      mrow[r] = mn;
      tsum[r] = 0.0f;
    }
    // P = exp(S - max) -> bf16 staging (per-wave LDS region) + row sums.
    __bf16* sPw = sP + w * 16 * 128;
#pragma unroll
    for (int j = 0; j < 8; ++j)
#pragma unroll
      for (int r = 0; r < 8; ++r) {
        float p = __expf(sacc[j][r] - mrow[r]);
        sPw[(r + 8 * half) * 128 + 16 * j + nc] = (__bf16)p;
        tsum[r] += p;
      }
#pragma unroll
    for (int r = 0; r < 8; ++r) {
      float s = tsum[r];
#pragma unroll
      for (int d = 1; d < 16; d <<= 1) s += __shfl_xor(s, d, 32);
      lrow[r] = lrow[r] * pscale[r] + s;
    }
#pragma unroll
    for (int j2 = 0; j2 < 8; ++j2)
#pragma unroll
      for (int r = 0; r < 8; ++r) oacc[j2][r] *= pscale[r];

    // O += P @ V (per-wave: same-wave LDS ordering makes sP safe to reread).
#pragma unroll
    for (int kk = 0; kk < 128; kk += 32) {
      v16bf a = frag_a(sPw + kk, lane);
#pragma unroll
      for (int j2 = 0; j2 < 8; ++j2) {
        v16bf bf = frag_b(sV + kk * 128 + 16 * j2, lane);
        oacc[j2] = wmma_bf16(a, bf, oacc[j2]);
      }
    }
  }

  // Normalize and store y[b][t][h*128+feat] bf16.
  float rinv[8];
#pragma unroll
  for (int r = 0; r < 8; ++r) rinv[r] = 1.0f / lrow[r];
#pragma unroll
  for (int j2 = 0; j2 < 8; ++j2)
#pragma unroll
    for (int r = 0; r < 8; ++r) {
      int t   = qRowBase + m0 + r + 8 * half;
      int col = h * 128 + 16 * j2 + nc;
      Y[(size_t)(b * TT + t) * (DK * HH) + col] = (__bf16)(oacc[j2][r] * rinv[r]);
    }
}

// ===========================================================================
// Kernel 3: out = y @ Wu + bu, f32 output. grid = (32,1).
// LDS: sY 128x128 bf16 + sWu 128x128 bf16 = 64 KB dynamic.
// ===========================================================================
__global__ __launch_bounds__(256) void out_proj_kernel(
    const __bf16* __restrict__ Y, const float* __restrict__ Wu,
    const float* __restrict__ bu, float* __restrict__ out) {
  extern __shared__ char smem[];
  __bf16* sY  = reinterpret_cast<__bf16*>(smem);
  __bf16* sWu = sY + 128 * 128;

  const int tid = threadIdx.x, lane = tid & 31, w = tid >> 5;
  const int rowBase = blockIdx.x * 128;
  const int mBase = (w >> 1) * 32, nBase = (w & 1) * 64;
  const int half = lane >> 4, nc = lane & 15;

  v8f acc[2][4] = {};
  for (int kb = 0; kb < DK * HH; kb += 128) {
    __syncthreads();
    // y tile (bf16): async global->LDS; rows rowBase..+128, cols kb..+128.
    {
      int row = tid >> 1, c8 = (tid & 1) << 6;        // two 64-elem halves/row
      const __bf16* src = Y + (size_t)(rowBase + row) * (DK * HH) + kb + c8;
      __bf16* dst = sY + row * 128 + c8;
#pragma unroll
      for (int i = 0; i < 8; ++i)
        async_copy_b128(dst + i * 8, src + i * 8);    // 8 bf16 = 16 B chunks
    }
    // Wu tile f32->bf16: rows kb..+128, all 128 cols.
    for (int i = tid; i < 4096; i += 256) {
      int kr = i >> 5, c4 = (i & 31) << 2;
      float4 f = *reinterpret_cast<const float4*>(Wu + (size_t)(kb + kr) * DK + c4);
      __bf16* d = &sWu[kr * 128 + c4];
      d[0] = (__bf16)f.x; d[1] = (__bf16)f.y; d[2] = (__bf16)f.z; d[3] = (__bf16)f.w;
    }
    async_wait0();
    __syncthreads();
#pragma unroll
    for (int kk = 0; kk < 128; kk += 32) {
      v16bf a0 = frag_a(sY + mBase * 128 + kk, lane);
      v16bf a1 = frag_a(sY + (mBase + 16) * 128 + kk, lane);
#pragma unroll
      for (int j = 0; j < 4; ++j) {
        v16bf bf = frag_b(sWu + kk * 128 + nBase + 16 * j, lane);
        acc[0][j] = wmma_bf16(a0, bf, acc[0][j]);
        acc[1][j] = wmma_bf16(a1, bf, acc[1][j]);
      }
    }
  }
#pragma unroll
  for (int i = 0; i < 2; ++i)
#pragma unroll
    for (int j = 0; j < 4; ++j)
#pragma unroll
      for (int r = 0; r < 8; ++r) {
        int row = rowBase + mBase + 16 * i + r + 8 * half;
        int col = nBase + 16 * j + nc;
        out[(size_t)row * DK + col] = acc[i][j][r] + bu[col];
      }
}

// ===========================================================================
extern "C" void kernel_launch(void* const* d_in, const int* in_sizes, int n_in,
                              void* d_out, int out_size, void* d_ws, size_t ws_size,
                              hipStream_t stream) {
  const float* x  = (const float*)d_in[0];
  const float* Wq = (const float*)d_in[1];
  const float* Wk = (const float*)d_in[2];
  const float* Wv = (const float*)d_in[3];
  const float* Wu = (const float*)d_in[4];
  const float* bu = (const float*)d_in[5];
  float* out = (float*)d_out;

  // Workspace: Q,K,V,Y in bf16 — 4 x (B*H*T*K)*2B = 32 MB total.
  const size_t elems = (size_t)BB * HH * TT * DK;   // 4,194,304
  char* ws = (char*)d_ws;
  __bf16* Qb = (__bf16*)(ws);
  __bf16* Kb = (__bf16*)(ws + 2 * elems);
  __bf16* Vb = (__bf16*)(ws + 4 * elems);
  __bf16* Yb = (__bf16*)(ws + 6 * elems);

  qkv_proj_kernel<<<dim3(MROWS / 128, 24), 256, 64 * 1024, stream>>>(
      x, Wq, Wk, Wv, Qb, Kb, Vb);
  flash_attn_kernel<<<dim3(TT / 128, BB * HH), 256, 128 * 1024, stream>>>(
      Qb, Kb, Vb, Yb);
  out_proj_kernel<<<dim3(MROWS / 128, 1), 256, 64 * 1024, stream>>>(
      Yb, Wu, bu, out);
}
